// GraphAttention_3418793967969
// MI455X (gfx1250) — matrix operand order repaired and checked by reading, not compiled, not run on hardware
//
#include <hip/hip_runtime.h>
#include <hip/hip_bf16.h>

typedef _Float16 half16 __attribute__((ext_vector_type(16)));
typedef float    f32x8  __attribute__((ext_vector_type(8)));

#define GAT_HEADS 4
#define GAT_OUTC  32
#define GAT_HC    128
#define GAT_INC   64
#define GAT_NEG   0.01f

// ---------------------------------------------------------------------------
// Kernel 0: init out1 (N*32) to 0, zmax (N*4) to -inf bits, denom (N*4) to 0
// ---------------------------------------------------------------------------
__global__ void gat_init_kernel(unsigned* zmax, float* denom, float* out1,
                                int n4, int n32) {
    int i = blockIdx.x * blockDim.x + threadIdx.x;
    int stride = gridDim.x * blockDim.x;
    for (int k = i; k < n4; k += stride) {
        zmax[k]  = 0xFF800000u;   // -inf
        denom[k] = 0.0f;
    }
    for (int k = i; k < n32; k += stride) out1[k] = 0.0f;
}

// ---------------------------------------------------------------------------
// Kernel 1: pre-swizzle the three 64x128 f32 weight matrices into per-lane
// f16 B-fragments (WMMA 16x16x32 B layout: lanes 0-15 -> N=lane, K=0..7,16..23
// packed 2-per-VGPR; lanes 16-31 -> N=lane-16, K=8..15,24..31).
// Fragment index: ((mat*8 + nt)*2 + ks)*32 + lane, 16 halves each.
// mat 0 = W_tgt (x_j), mat 1 = W_src (x_i), mat 2 = W_edge.
// ---------------------------------------------------------------------------
__global__ void gat_prep_weights(const float* __restrict__ Wt,
                                 const float* __restrict__ Ws,
                                 const float* __restrict__ We,
                                 _Float16* __restrict__ wf) {
    int idx = blockIdx.x * blockDim.x + threadIdx.x;  // 3*8*2*32 = 1536 items
    if (idx >= 3 * 8 * 2 * 32) return;
    int lane = idx & 31;
    int ks   = (idx >> 5) & 1;
    int nt   = (idx >> 6) & 7;
    int mat  = idx >> 9;
    const float* W = (mat == 0) ? Wt : ((mat == 1) ? Ws : We);
    int col   = lane & 15;
    int basek = (lane < 16) ? 0 : 8;
    _Float16* dst = wf + (size_t)idx * 16;
#pragma unroll
    for (int i = 0; i < 16; ++i)
        dst[i] = (_Float16)W[(size_t)(ks * 32 + basek + i + (i >= 8 ? 8 : 0)) * GAT_HC + nt * 16 + col];
}

// ---------------------------------------------------------------------------
// A-fragment builder: 16-bit A 16x32 layout. lane<16: halves 0..7 = K 0..7,
// halves 8..15 = K 16..23 ; lane>=16: +8 on both ranges. Row = lane&15.
// ---------------------------------------------------------------------------
__device__ inline half16 gat_load_a(const float* __restrict__ x, int rowBase,
                                    int lane, int ks, int E) {
    int m   = lane & 15;
    int row = rowBase + m;
    if (row >= E) row = E - 1;                 // clamp (tail tiles)
    int c0 = ks * 32 + ((lane >= 16) ? 8 : 0); // multiple of 8 -> 32B aligned
    const float4* p = (const float4*)(x + (size_t)row * GAT_INC + c0);
    float4 f0 = p[0];
    float4 f1 = p[1];
    const float4* q = (const float4*)(x + (size_t)row * GAT_INC + c0 + 16);
    float4 f2 = q[0];
    float4 f3 = q[1];
    half16 a;
    a[0]  = (_Float16)f0.x; a[1]  = (_Float16)f0.y;
    a[2]  = (_Float16)f0.z; a[3]  = (_Float16)f0.w;
    a[4]  = (_Float16)f1.x; a[5]  = (_Float16)f1.y;
    a[6]  = (_Float16)f1.z; a[7]  = (_Float16)f1.w;
    a[8]  = (_Float16)f2.x; a[9]  = (_Float16)f2.y;
    a[10] = (_Float16)f2.z; a[11] = (_Float16)f2.w;
    a[12] = (_Float16)f3.x; a[13] = (_Float16)f3.y;
    a[14] = (_Float16)f3.z; a[15] = (_Float16)f3.w;
    return a;
}

// ---------------------------------------------------------------------------
// Kernel 2: fused triple-GEMM per 16-edge tile (one wave each).
//   acc  = x_j @ W_tgt            -> store T (f16) for message pass
//   acc += x_i @ W_src + ea @ W_edge
//   z[e][h] = sum_c leakyrelu(acc)[h*32+c] * attn[h*32+c]
// 48 v_wmma_f32_16x16x32_f16 per tile. wave32, 8 waves / 256-thread block.
// ---------------------------------------------------------------------------
__global__ void __launch_bounds__(256) gat_gemm_kernel(
    const float* __restrict__ xi, const float* __restrict__ xj,
    const float* __restrict__ ea, const float* __restrict__ attn,
    const half16* __restrict__ wfrag, _Float16* __restrict__ t_h,
    float* __restrict__ z, int E, int ntiles) {
    int wave = threadIdx.x >> 5;
    int lane = threadIdx.x & 31;
    int tile = blockIdx.x * 8 + wave;
    if (tile >= ntiles) return;   // uniform per wave
    int base = tile * 16;

    f32x8 acc[8];
    f32x8 zero = {};
#pragma unroll
    for (int nt = 0; nt < 8; ++nt) acc[nt] = zero;

    // ---- T = x_j @ W_tgt ----
#pragma unroll
    for (int ks = 0; ks < 2; ++ks) {
        half16 a = gat_load_a(xj, base, lane, ks, E);
#pragma unroll
        for (int nt = 0; nt < 8; ++nt) {
            half16 b = wfrag[(size_t)((0 * 8 + nt) * 2 + ks) * 32 + lane];
            acc[nt] = __builtin_amdgcn_wmma_f32_16x16x32_f16(
                false, a, false, b, (short)0, acc[nt], false, false);
        }
    }

    // C/D layout: VGPR r on lane l -> row m = r + (l>=16 ? 8 : 0), col = l&15
    int col = lane & 15;
    int m0  = (lane >= 16) ? 8 : 0;
#pragma unroll
    for (int nt = 0; nt < 8; ++nt) {
#pragma unroll
        for (int r = 0; r < 8; ++r) {
            int row = base + m0 + r;
            if (row < E)
                t_h[(size_t)row * GAT_HC + nt * 16 + col] = (_Float16)acc[nt][r];
        }
    }

    // ---- acc += x_i @ W_src ----
#pragma unroll
    for (int ks = 0; ks < 2; ++ks) {
        half16 a = gat_load_a(xi, base, lane, ks, E);
#pragma unroll
        for (int nt = 0; nt < 8; ++nt) {
            half16 b = wfrag[(size_t)((1 * 8 + nt) * 2 + ks) * 32 + lane];
            acc[nt] = __builtin_amdgcn_wmma_f32_16x16x32_f16(
                false, a, false, b, (short)0, acc[nt], false, false);
        }
    }
    // ---- acc += edge_attr @ W_edge ----
#pragma unroll
    for (int ks = 0; ks < 2; ++ks) {
        half16 a = gat_load_a(ea, base, lane, ks, E);
#pragma unroll
        for (int nt = 0; nt < 8; ++nt) {
            half16 b = wfrag[(size_t)((2 * 8 + nt) * 2 + ks) * 32 + lane];
            acc[nt] = __builtin_amdgcn_wmma_f32_16x16x32_f16(
                false, a, false, b, (short)0, acc[nt], false, false);
        }
    }

    // ---- attention logits: z[m][h] = sum_n lrelu(acc[m][n]) * attn[n] ----
    // attn flat [H*C] index == column index n.
    float attnv[8];
#pragma unroll
    for (int nt = 0; nt < 8; ++nt) attnv[nt] = attn[nt * 16 + col];

#pragma unroll
    for (int h = 0; h < 4; ++h) {
        float ph[8];
#pragma unroll
        for (int r = 0; r < 8; ++r) {
            float x0 = acc[2 * h][r];
            x0 = (x0 > 0.0f) ? x0 : GAT_NEG * x0;
            float x1 = acc[2 * h + 1][r];
            x1 = (x1 > 0.0f) ? x1 : GAT_NEG * x1;
            ph[r] = x0 * attnv[2 * h] + x1 * attnv[2 * h + 1];
        }
        // xor-reduce across the 16-lane half-group (stays within group)
#pragma unroll
        for (int mask = 1; mask < 16; mask <<= 1) {
#pragma unroll
            for (int r = 0; r < 8; ++r)
                ph[r] += __shfl_xor(ph[r], mask, 32);
        }
        if (col == 0) {
#pragma unroll
            for (int r = 0; r < 8; ++r) {
                int row = base + m0 + r;
                if (row < E) z[(size_t)row * GAT_HEADS + h] = ph[r];
            }
        }
    }
}

// ---------------------------------------------------------------------------
// Kernel 3: segment max over senders (sign-split int atomic trick).
// ---------------------------------------------------------------------------
__global__ void gat_seg_max(const float* __restrict__ z,
                            const int* __restrict__ senders,
                            unsigned* __restrict__ zmax, int EH) {
    int i = blockIdx.x * blockDim.x + threadIdx.x;
    if (i >= EH) return;
    int e = i >> 2, h = i & 3;
    int s = senders[e];
    float v = z[i];
    unsigned* addr = &zmax[(size_t)s * GAT_HEADS + h];
    if (v >= 0.0f) atomicMax((int*)addr, __float_as_int(v));
    else           atomicMin(addr, __float_as_uint(v));
}

// ---------------------------------------------------------------------------
// Kernel 4: ez = exp(z - zmax[seg]) ; denom[seg] += ez
// ---------------------------------------------------------------------------
__global__ void gat_seg_expsum(const float* __restrict__ z,
                               const int* __restrict__ senders,
                               const unsigned* __restrict__ zmax,
                               float* __restrict__ ez,
                               float* __restrict__ denom, int EH) {
    int i = blockIdx.x * blockDim.x + threadIdx.x;
    if (i >= EH) return;
    int e = i >> 2, h = i & 3;
    int s = senders[e];
    float mx = __uint_as_float(zmax[(size_t)s * GAT_HEADS + h]);
    float v  = expf(z[i] - mx);
    ez[i] = v;
    atomicAdd(&denom[(size_t)s * GAT_HEADS + h], v);
}

// ---------------------------------------------------------------------------
// Kernel 5: out2[e][c] = 0.25 * sum_h t[e][h*32+c] * a[e][h]  (== m.mean(1))
//           out1[recv[e]][c] += out2[e][c]                    (== aggr.mean(1))
// 32 threads per edge (one wave handles one edge per lane-group of 32).
// ---------------------------------------------------------------------------
__global__ void gat_finalize(const _Float16* __restrict__ t_h,
                             const float* __restrict__ ez,
                             const float* __restrict__ denom,
                             const int* __restrict__ senders,
                             const int* __restrict__ receivers,
                             float* __restrict__ out1,
                             float* __restrict__ out2, int E) {
    long long idx = (long long)blockIdx.x * blockDim.x + threadIdx.x;
    if (idx >= (long long)E * 32) return;
    int e = (int)(idx >> 5);
    int c = (int)(idx & 31);
    int s   = senders[e];
    int rcv = receivers[e];
    float v = 0.0f;
#pragma unroll
    for (int h = 0; h < 4; ++h) {
        float a = ez[(size_t)e * 4 + h] / denom[(size_t)s * 4 + h];
        v += (float)t_h[(size_t)e * GAT_HC + h * 32 + c] * a;
    }
    v *= 0.25f;
    out2[(size_t)e * 32 + c] = v;
    atomicAdd(&out1[(size_t)rcv * 32 + c], v);
}

// ---------------------------------------------------------------------------
extern "C" void kernel_launch(void* const* d_in, const int* in_sizes, int n_in,
                              void* d_out, int out_size, void* d_ws, size_t ws_size,
                              hipStream_t stream) {
    const float* xi      = (const float*)d_in[0];
    const float* xj      = (const float*)d_in[1];
    const float* ea      = (const float*)d_in[2];
    const int*   senders = (const int*)d_in[3];
    const int*   recvs   = (const int*)d_in[4];
    // d_in[5] = num_nodes (device scalar, unused — N derived from out_size)
    const float* Wsrc    = (const float*)d_in[6];
    const float* Wtgt    = (const float*)d_in[7];
    const float* Wedge   = (const float*)d_in[8];
    const float* attn    = (const float*)d_in[9];

    int E = in_sizes[3];
    int N = (int)(((long long)out_size - (long long)E * 32) / 32);

    // workspace layout (256B aligned slices)
    char* ws = (char*)d_ws;
    size_t off = 0;
    auto take = [&](size_t bytes) -> void* {
        off = (off + 255) & ~(size_t)255;
        void* p = ws + off;
        off += bytes;
        return p;
    };
    _Float16* wf    = (_Float16*)take((size_t)3 * 8 * 2 * 32 * 16 * sizeof(_Float16));
    _Float16* t_h   = (_Float16*)take((size_t)E * GAT_HC * sizeof(_Float16));
    float*    z     = (float*)take((size_t)E * GAT_HEADS * sizeof(float));
    float*    ezb   = (float*)take((size_t)E * GAT_HEADS * sizeof(float));
    unsigned* zmax  = (unsigned*)take((size_t)N * GAT_HEADS * sizeof(unsigned));
    float*    denom = (float*)take((size_t)N * GAT_HEADS * sizeof(float));
    (void)ws_size; (void)n_in;

    float* out1 = (float*)d_out;             // [N, 32]
    float* out2 = out1 + (size_t)N * 32;     // [E, 32]

    gat_init_kernel<<<512, 256, 0, stream>>>(zmax, denom, out1, N * 4, N * 32);
    gat_prep_weights<<<6, 256, 0, stream>>>(Wtgt, Wsrc, Wedge, wf);

    int ntiles = (E + 15) / 16;
    int blocks = (ntiles + 7) / 8;   // 8 waves (tiles) per 256-thread block
    gat_gemm_kernel<<<blocks, 256, 0, stream>>>(xi, xj, ea, attn,
                                                (const half16*)wf, t_h, z, E, ntiles);

    int EH = E * 4;
    gat_seg_max<<<(EH + 255) / 256, 256, 0, stream>>>(z, senders, zmax, EH);
    gat_seg_expsum<<<(EH + 255) / 256, 256, 0, stream>>>(z, senders, zmax, ezb, denom, EH);

    long long tot = (long long)E * 32;
    gat_finalize<<<(int)((tot + 255) / 256), 256, 0, stream>>>(
        t_h, ezb, denom, senders, recvs, out1, out2, E);
}
